// MambaEncoderDecoder_48859547959721
// MI455X (gfx1250) — compile-verified, matrix operand order checked by reading, not run on hardware
//
#include <hip/hip_runtime.h>
#include <hip/hip_bf16.h>
#include <math.h>

// ---------------------------------------------------------------------------
// MambaEncoderDecoder for MI455X (gfx1250, wave32).
// GEMMs -> v_wmma_f32_16x16x32_f16 (f16 weights, f32 accum), K templated so the
//          k-loop fully unrolls; A/B tiles fetched as b128 chunks.
// Expander (K=4) -> v_wmma_f32_16x16x4_f32.
// Scan -> f32 registers; B/C/dt staged to LDS via GLOBAL_LOAD_ASYNC_TO_LDS_B32
//         (ASYNCcnt) when the builtin exists, plain LDS stores otherwise.
// ---------------------------------------------------------------------------

#define DM 64
#define DI 128
#define DS_N 16
#define DC 4
#define R_RANK 4
#define NHEAD 4
#define NL 3
#define B_READS 128
#define L_READ 512
#define N_SEQ 96
#define TOK (B_READS * L_READ)   // 65536

typedef __attribute__((ext_vector_type(16))) _Float16 v16h;
typedef __attribute__((ext_vector_type(8)))  _Float16 v8h;
typedef __attribute__((ext_vector_type(8)))  float    v8f;
typedef __attribute__((ext_vector_type(2)))  float    v2f;

#if defined(__has_builtin)
#if __has_builtin(__builtin_amdgcn_global_load_async_to_lds_b32) && \
    __has_builtin(__builtin_amdgcn_s_wait_asynccnt)
#define HAVE_ASYNC_LDS 1
#endif
#endif

#ifdef HAVE_ASYNC_LDS
typedef __attribute__((address_space(1))) int as1_int;
typedef __attribute__((address_space(3))) int as3_int;
#endif

__device__ __forceinline__ float silu_f(float v) {
    return v / (1.0f + __expf(-v));
}
__device__ __forceinline__ float softplus_f(float v) {
    return (v > 20.0f) ? v : log1pf(__expf(v));
}

// ---------------------------------------------------------------------------
// Weight f32 -> f16 conversion (one-shot prep; x_proj padded 36 -> 48 rows)
// ---------------------------------------------------------------------------
__global__ void prep_f16_kernel(const float* __restrict__ in_proj_w,   // (NL,256,64)
                                const float* __restrict__ x_proj_w,    // (NL,36,128)
                                const float* __restrict__ out_w,       // (NL,64,128)
                                _Float16* __restrict__ w_in,           // (NL,256,64)
                                _Float16* __restrict__ w_xp,           // (NL,48,128) padded
                                _Float16* __restrict__ w_out)          // (NL,64,128)
{
    const int n_in  = NL * 2 * DI * DM;   // 49152
    const int n_xp  = NL * 48 * DI;       // 18432
    const int n_out = NL * DM * DI;       // 24576
    int i = blockIdx.x * blockDim.x + threadIdx.x;
    if (i < n_in) {
        w_in[i] = (_Float16)in_proj_w[i];
    } else if (i < n_in + n_xp) {
        int j = i - n_in;
        int layer = j / (48 * DI);
        int row   = (j / DI) % 48;
        int col   = j % DI;
        w_xp[j] = (row < (R_RANK + 2 * DS_N))
                    ? (_Float16)x_proj_w[(layer * (R_RANK + 2 * DS_N) + row) * DI + col]
                    : (_Float16)0.0f;
    } else if (i < n_in + n_xp + n_out) {
        int j = i - n_in - n_xp;
        w_out[j] = (_Float16)out_w[j];
    }
}

// ---------------------------------------------------------------------------
// Expander: H(M,64) = reads(M,4) @ ew(64,4)^T + eb.  K=4 == WMMA f32 16x16x4.
// ---------------------------------------------------------------------------
__global__ __launch_bounds__(256)
void expander_kernel(const float* __restrict__ reads,
                     const float* __restrict__ ew,
                     const float* __restrict__ eb,
                     float* __restrict__ H)
{
    int wave = (blockIdx.x * blockDim.x + threadIdx.x) >> 5;
    int lane = threadIdx.x & 31;
    int mt = wave >> 2;          // 64/16 = 4 n-tiles
    int nt = wave & 3;
    if (mt * 16 >= TOK) return;  // wave-uniform
    int hi  = lane >> 4;
    int l15 = lane & 15;

#if __has_builtin(__builtin_amdgcn_wmma_f32_16x16x4_f32)
    v2f a, b;
    const float* ar = reads + (size_t)(mt * 16 + l15) * 4 + hi * 2;
    a[0] = ar[0]; a[1] = ar[1];
    const float* br = ew + (size_t)(nt * 16 + l15) * 4 + hi * 2;
    b[0] = br[0]; b[1] = br[1];
    v8f acc = {};
    acc = __builtin_amdgcn_wmma_f32_16x16x4_f32(false, a, false, b,
                                                (short)0, acc, false, false);
    #pragma unroll
    for (int r = 0; r < 8; ++r) {
        int row = mt * 16 + r + (hi << 3);
        int col = nt * 16 + l15;
        H[(size_t)row * DM + col] = acc[r] + eb[col];
    }
#else
    #pragma unroll
    for (int r = 0; r < 8; ++r) {
        int row = mt * 16 + r + (hi << 3);
        int col = nt * 16 + l15;
        const float* ar = reads + (size_t)row * 4;
        const float* wr = ew + (size_t)col * 4;
        float s = eb[col];
        #pragma unroll
        for (int k = 0; k < 4; ++k) s += ar[k] * wr[k];
        H[(size_t)row * DM + col] = s;
    }
#endif
}

// ---------------------------------------------------------------------------
// Generic GEMM: C(M,N) = A(M,KK,f32) @ W(N,KK,f16)^T  with f32 accumulate.
// KK template param (64/128) -> fully unrolled k-loop.
// One wave per 16x16 output tile, v_wmma_f32_16x16x32_f16.
// A-tile packing per ISA: lane element i -> k = (i&7) + 16*(i>=8) + 8*(lane>=16),
// i.e. two contiguous 8-element runs per lane: [8*hi, 8*hi+8) and [16+8*hi, +8).
// ---------------------------------------------------------------------------
template <int KK>
__global__ __launch_bounds__(256)
void gemm_f16_kernel(const float* __restrict__ A,
                     const _Float16* __restrict__ W,
                     float* __restrict__ C,
                     int M, int N)
{
    int wave = (blockIdx.x * blockDim.x + threadIdx.x) >> 5;
    int lane = threadIdx.x & 31;
    int ntiles = N >> 4;
    int mtiles = M >> 4;
    if (wave >= mtiles * ntiles) return;     // wave-uniform
    int mt = wave / ntiles;
    int nt = wave % ntiles;
    int hi  = lane >> 4;
    int l15 = lane & 15;

    const float*    arow = A + (size_t)(mt * 16 + l15) * KK;
    const _Float16* brow = W + (size_t)(nt * 16 + l15) * KK;

    v8f acc = {};
    #pragma unroll
    for (int k0 = 0; k0 < KK; k0 += 32) {
        // A: two contiguous float4-pairs (rows are >=256B aligned, offsets 32B mult.)
        const float4* ap0 = (const float4*)(arow + k0 + hi * 8);
        const float4* ap1 = (const float4*)(arow + k0 + 16 + hi * 8);
        float4 a0 = ap0[0], a1 = ap0[1];
        float4 a2 = ap1[0], a3 = ap1[1];
        // B: two contiguous 8-half (16B) chunks
        const v8h* bp0 = (const v8h*)(brow + k0 + hi * 8);
        const v8h* bp1 = (const v8h*)(brow + k0 + 16 + hi * 8);
        v8h b0 = bp0[0], b1 = bp1[0];

        v16h av, bv;
        av[0] = (_Float16)a0.x; av[1] = (_Float16)a0.y;
        av[2] = (_Float16)a0.z; av[3] = (_Float16)a0.w;
        av[4] = (_Float16)a1.x; av[5] = (_Float16)a1.y;
        av[6] = (_Float16)a1.z; av[7] = (_Float16)a1.w;
        av[8]  = (_Float16)a2.x; av[9]  = (_Float16)a2.y;
        av[10] = (_Float16)a2.z; av[11] = (_Float16)a2.w;
        av[12] = (_Float16)a3.x; av[13] = (_Float16)a3.y;
        av[14] = (_Float16)a3.z; av[15] = (_Float16)a3.w;
        #pragma unroll
        for (int i = 0; i < 8; ++i) { bv[i] = b0[i]; bv[i + 8] = b1[i]; }

        acc = __builtin_amdgcn_wmma_f32_16x16x32_f16(false, av, false, bv,
                                                     (short)0, acc, false, false);
    }
    #pragma unroll
    for (int r = 0; r < 8; ++r) {
        int row = mt * 16 + r + (hi << 3);
        int col = nt * 16 + l15;
        C[(size_t)row * N + col] = acc[r];
    }
}

// ---------------------------------------------------------------------------
// Depthwise causal conv (DC=4) + bias + SiLU.  x = xz[..., 0:128].
// ---------------------------------------------------------------------------
__global__ void conv_silu_kernel(const float* __restrict__ xz,   // (B,L,256)
                                 const float* __restrict__ cw,   // (128,1,4) layer slice
                                 const float* __restrict__ cb,   // (128)
                                 float* __restrict__ xS)         // (B,L,128)
{
    int idx = blockIdx.x * blockDim.x + threadIdx.x;
    if (idx >= TOK * DI) return;
    int d = idx & (DI - 1);
    int l = (idx >> 7) & (L_READ - 1);
    int b = idx >> 16;
    float s = cb[d];
    #pragma unroll
    for (int t = 0; t < DC; ++t) {
        int ll = l - (DC - 1) + t;
        if (ll >= 0)
            s += xz[((size_t)(b * L_READ + ll)) * (2 * DI) + d] * cw[d * DC + t];
    }
    xS[idx] = silu_f(s);
}

// ---------------------------------------------------------------------------
// Selective scan.  One block per batch row, 128 threads = channels.
// xd row (stride 48): [0:4]=dt_raw inputs, [4:20]=B_t, [20:36]=C_t.
// Per-step 48-float broadcast staged via async global->LDS when available.
// ---------------------------------------------------------------------------
__global__ __launch_bounds__(128)
void scan_kernel(const float* __restrict__ xS,     // (B,L,128)
                 const float* __restrict__ xd,     // (B*L,48)
                 const float* __restrict__ dtw,    // (128,4) layer slice
                 const float* __restrict__ dtb,    // (128)
                 const float* __restrict__ A_log,  // (128,16) layer slice
                 float* __restrict__ Y)            // (B,L,128)
{
    __shared__ float sBC[48];                       // [0:4]=dt in, [4:20]=B, [20:36]=C
    int b = blockIdx.x;
    int d = threadIdx.x;

    float w0 = dtw[d * 4 + 0], w1 = dtw[d * 4 + 1];
    float w2 = dtw[d * 4 + 2], w3 = dtw[d * 4 + 3];
    float db = dtb[d];

    float a[DS_N];
    #pragma unroll
    for (int n = 0; n < DS_N; ++n) a[n] = -__expf(A_log[d * DS_N + n]);

    float h[DS_N];
    #pragma unroll
    for (int n = 0; n < DS_N; ++n) h[n] = 0.0f;

    for (int l = 0; l < L_READ; ++l) {
        size_t t = (size_t)b * L_READ + l;
        __syncthreads();
#ifdef HAVE_ASYNC_LDS
        if (d < 48) {
            // generic-shared low 32 bits == LDS byte offset (aperture truncation);
            // global generic == AS1 addresses.
            __builtin_amdgcn_global_load_async_to_lds_b32(
                (as1_int*)(unsigned long long)(const void*)(xd + t * 48 + d),
                (as3_int*)(unsigned int)(unsigned long long)(const void*)&sBC[d],
                0, 0);
        }
        __builtin_amdgcn_s_wait_asynccnt(0);
#else
        if (d < 48) sBC[d] = xd[t * 48 + d];
#endif
        if (d == 49 && l + 1 < L_READ)
            __builtin_prefetch(&xd[(t + 1) * 48], 0, 0);   // global_prefetch_b8
        __syncthreads();

        float dt = softplus_f(sBC[0] * w0 + sBC[1] * w1 + sBC[2] * w2 + sBC[3] * w3 + db);
        float dx = dt * xS[t * DI + d];
        float y = 0.0f;
        #pragma unroll
        for (int n = 0; n < DS_N; ++n) {
            h[n] = __expf(dt * a[n]) * h[n] + dx * sBC[R_RANK + n];
            y += h[n] * sBC[R_RANK + DS_N + n];
        }
        Y[t * DI + d] = y;
    }
}

// ---------------------------------------------------------------------------
// Gating (in place on Y):  Y = (Y + xS * D) * silu(z),  z = xz[...,128:256]
// ---------------------------------------------------------------------------
__global__ void gate_kernel(float* __restrict__ Y,
                            const float* __restrict__ xS,
                            const float* __restrict__ xz,
                            const float* __restrict__ Dp)
{
    int idx = blockIdx.x * blockDim.x + threadIdx.x;
    if (idx >= TOK * DI) return;
    int d = idx & (DI - 1);
    size_t t = (size_t)(idx >> 7);
    float z = xz[t * (2 * DI) + DI + d];
    Y[idx] = (Y[idx] + xS[idx] * Dp[d]) * silu_f(z);
}

// ---------------------------------------------------------------------------
// LayerNorm(64) + ReLU, one thread per token, float4 row traffic (rows 256B aligned)
// ---------------------------------------------------------------------------
__global__ void ln_relu_kernel(const float* __restrict__ X,   // (TOK,64)
                               const float* __restrict__ w,
                               const float* __restrict__ bb,
                               float* __restrict__ Hn)        // (TOK,64)
{
    int t = blockIdx.x * blockDim.x + threadIdx.x;
    if (t >= TOK) return;
    const float4* row = (const float4*)(X + (size_t)t * DM);
    float m = 0.0f, v = 0.0f;
    #pragma unroll
    for (int i = 0; i < DM / 4; ++i) {
        float4 r = row[i];
        m += r.x + r.y + r.z + r.w;
        v += r.x * r.x + r.y * r.y + r.z * r.z + r.w * r.w;
    }
    m *= (1.0f / DM);
    v = v * (1.0f / DM) - m * m;          // E[x^2]-E[x]^2
    float inv = rsqrtf(fmaxf(v, 0.0f) + 1e-5f);
    float4* out = (float4*)(Hn + (size_t)t * DM);
    const float4* wv = (const float4*)w;
    const float4* bv = (const float4*)bb;
    #pragma unroll
    for (int i = 0; i < DM / 4; ++i) {
        float4 r = row[i], wr = wv[i], br = bv[i], o;
        o.x = fmaxf((r.x - m) * inv * wr.x + br.x, 0.0f);
        o.y = fmaxf((r.y - m) * inv * wr.y + br.y, 0.0f);
        o.z = fmaxf((r.z - m) * inv * wr.z + br.z, 0.0f);
        o.w = fmaxf((r.w - m) * inv * wr.w + br.w, 0.0f);
        out[i] = o;
    }
}

// ---------------------------------------------------------------------------
// Pooling: enc[b,0:64]=mean_l h, enc[b,64:128]=h[b,L-1,:]
// ---------------------------------------------------------------------------
__global__ void pool_kernel(const float* __restrict__ Hf, float* __restrict__ enc)
{
    int b = blockIdx.x;
    int d = threadIdx.x;          // 64
    float s = 0.0f;
    for (int l = 0; l < L_READ; ++l)
        s += Hf[((size_t)(b * L_READ + l)) * DM + d];
    enc[b * (2 * DM) + d] = s * (1.0f / L_READ);
    enc[b * (2 * DM) + DM + d] = Hf[((size_t)(b * L_READ + L_READ - 1)) * DM + d];
}

// ---------------------------------------------------------------------------
// Q/K projections (tiny: 224 x 64 outputs, K=128)
// ---------------------------------------------------------------------------
__global__ void qk_kernel(const float* __restrict__ enc,
                          const int* __restrict__ sidx,
                          const float* __restrict__ qw, const float* __restrict__ qb,
                          const float* __restrict__ kw, const float* __restrict__ kb,
                          float* __restrict__ Q, float* __restrict__ Kc)
{
    int i = blockIdx.x * blockDim.x + threadIdx.x;
    if (i < N_SEQ * DM) {
        int r = i / DM, c = i % DM;
        const float* e = enc + (size_t)sidx[r] * (2 * DM);
        float s = qb[c];
        for (int k = 0; k < 2 * DM; ++k) s += e[k] * qw[c * (2 * DM) + k];
        Q[i] = s;
    } else if (i < (N_SEQ + B_READS) * DM) {
        int j = i - N_SEQ * DM;
        int r = j / DM, c = j % DM;
        const float* e = enc + (size_t)r * (2 * DM);
        float s = kb[c];
        for (int k = 0; k < 2 * DM; ++k) s += e[k] * kw[c * (2 * DM) + k];
        Kc[j] = s;
    }
}

// ---------------------------------------------------------------------------
// Scores: per (q,k): s_h = q_h . k_h ; out = max_o ( relu(s) . mix_w[o] + mix_b[o] )
// ---------------------------------------------------------------------------
__global__ void score_kernel(const float* __restrict__ Q,
                             const float* __restrict__ Kc,
                             const float* __restrict__ mw,
                             const float* __restrict__ mb,
                             float* __restrict__ out)
{
    int i = blockIdx.x * blockDim.x + threadIdx.x;
    if (i >= N_SEQ * B_READS) return;
    int qi = i / B_READS, ki = i % B_READS;
    float s[NHEAD];
    #pragma unroll
    for (int hh = 0; hh < NHEAD; ++hh) {
        float t = 0.0f;
        #pragma unroll
        for (int d = 0; d < DM / NHEAD; ++d)
            t += Q[qi * DM + hh * (DM / NHEAD) + d] * Kc[ki * DM + hh * (DM / NHEAD) + d];
        s[hh] = fmaxf(t, 0.0f);
    }
    float best = -INFINITY;
    #pragma unroll
    for (int o = 0; o < NHEAD; ++o) {
        float m = mb[o];
        #pragma unroll
        for (int hh = 0; hh < NHEAD; ++hh) m += s[hh] * mw[o * NHEAD + hh];
        best = fmaxf(best, m);
    }
    out[i] = best;
}

// ---------------------------------------------------------------------------
// Launcher
// ---------------------------------------------------------------------------
extern "C" void kernel_launch(void* const* d_in, const int* in_sizes, int n_in,
                              void* d_out, int out_size, void* d_ws, size_t ws_size,
                              hipStream_t stream)
{
    (void)in_sizes; (void)n_in; (void)out_size; (void)ws_size;

    const float* reads  = (const float*)d_in[0];
    const int*   sidx   = (const int*)  d_in[1];
    const float* ew     = (const float*)d_in[2];
    const float* eb     = (const float*)d_in[3];
    const float* inw    = (const float*)d_in[4];
    const float* cw     = (const float*)d_in[5];
    const float* cb     = (const float*)d_in[6];
    const float* xpw    = (const float*)d_in[7];
    const float* dtw    = (const float*)d_in[8];
    const float* dtb    = (const float*)d_in[9];
    const float* A_log  = (const float*)d_in[10];
    const float* Dp     = (const float*)d_in[11];
    const float* ow     = (const float*)d_in[12];
    const float* lnw    = (const float*)d_in[13];
    const float* lnb    = (const float*)d_in[14];
    const float* qw     = (const float*)d_in[15];
    const float* qb     = (const float*)d_in[16];
    const float* kw     = (const float*)d_in[17];
    const float* kb     = (const float*)d_in[18];
    const float* mw     = (const float*)d_in[19];
    const float* mb     = (const float*)d_in[20];

    char* ws = (char*)d_ws;
    size_t off = 0;
    auto take = [&](size_t bytes) { size_t o = off; off += (bytes + 255) & ~(size_t)255; return o; };
    float*    H    = (float*)(ws + take((size_t)TOK * DM * 4));
    float*    XZ   = (float*)(ws + take((size_t)TOK * 2 * DI * 4));
    float*    XS   = (float*)(ws + take((size_t)TOK * DI * 4));
    float*    XD   = (float*)(ws + take((size_t)TOK * 48 * 4));
    float*    Y    = (float*)(ws + take((size_t)TOK * DI * 4));
    float*    HO   = (float*)(ws + take((size_t)TOK * DM * 4));
    float*    ENC  = (float*)(ws + take((size_t)B_READS * 2 * DM * 4));
    float*    Q    = (float*)(ws + take((size_t)N_SEQ * DM * 4));
    float*    Kc   = (float*)(ws + take((size_t)B_READS * DM * 4));
    _Float16* W_IN = (_Float16*)(ws + take((size_t)NL * 2 * DI * DM * 2));
    _Float16* W_XP = (_Float16*)(ws + take((size_t)NL * 48 * DI * 2));
    _Float16* W_OUT= (_Float16*)(ws + take((size_t)NL * DM * DI * 2));

    // 1. weight conversion
    {
        int n = NL * 2 * DI * DM + NL * 48 * DI + NL * DM * DI;
        prep_f16_kernel<<<(n + 255) / 256, 256, 0, stream>>>(inw, xpw, ow, W_IN, W_XP, W_OUT);
    }
    // 2. expander -> H
    {
        int waves = (TOK / 16) * (DM / 16);           // 16384
        expander_kernel<<<(waves * 32 + 255) / 256, 256, 0, stream>>>(reads, ew, eb, H);
    }
    // 3. mamba layers
    for (int layer = 0; layer < NL; ++layer) {
        // in_proj: XZ = H @ inw^T   (M=TOK, N=256, K=64)
        {
            int waves = (TOK / 16) * (2 * DI / 16);   // 65536
            gemm_f16_kernel<DM><<<(waves * 32 + 255) / 256, 256, 0, stream>>>(
                H, W_IN + (size_t)layer * 2 * DI * DM, XZ, TOK, 2 * DI);
        }
        // conv + silu
        {
            int n = TOK * DI;
            conv_silu_kernel<<<(n + 255) / 256, 256, 0, stream>>>(
                XZ, cw + (size_t)layer * DI * DC, cb + (size_t)layer * DI, XS);
        }
        // x_proj: XD = XS @ xpw^T  (N padded 36->48, K=128)
        {
            int waves = (TOK / 16) * (48 / 16);       // 12288
            gemm_f16_kernel<DI><<<(waves * 32 + 255) / 256, 256, 0, stream>>>(
                XS, W_XP + (size_t)layer * 48 * DI, XD, TOK, 48);
        }
        // selective scan -> Y
        scan_kernel<<<B_READS, DI, 0, stream>>>(
            XS, XD,
            dtw + (size_t)layer * DI * R_RANK, dtb + (size_t)layer * DI,
            A_log + (size_t)layer * DI * DS_N, Y);
        // gating in place
        {
            int n = TOK * DI;
            gate_kernel<<<(n + 255) / 256, 256, 0, stream>>>(Y, XS, XZ, Dp + (size_t)layer * DI);
        }
        // out_proj: HO = Y @ ow^T  (M=TOK, N=64, K=128)
        {
            int waves = (TOK / 16) * (DM / 16);       // 16384
            gemm_f16_kernel<DI><<<(waves * 32 + 255) / 256, 256, 0, stream>>>(
                Y, W_OUT + (size_t)layer * DM * DI, HO, TOK, DM);
        }
        // layernorm + relu -> H
        ln_relu_kernel<<<(TOK + 255) / 256, 256, 0, stream>>>(
            HO, lnw + (size_t)layer * DM, lnb + (size_t)layer * DM, H);
    }
    // 4. pooling
    pool_kernel<<<B_READS, DM, 0, stream>>>(H, ENC);
    // 5. q/k projections
    {
        int n = (N_SEQ + B_READS) * DM;
        qk_kernel<<<(n + 255) / 256, 256, 0, stream>>>(ENC, sidx, qw, qb, kw, kb, Q, Kc);
    }
    // 6. scores -> out (96,128)
    {
        int n = N_SEQ * B_READS;
        score_kernel<<<(n + 255) / 256, 256, 0, stream>>>(Q, Kc, mw, mb, (float*)d_out);
    }
}
